// MultiHeadAttention_78262894067878
// MI455X (gfx1250) — matrix operand order repaired and checked by reading, not compiled, not run on hardware
//
#include <hip/hip_runtime.h>

// ---------------------------------------------------------------------------
// MI455X (gfx1250) multi-head attention, bf16 WMMA with f32 accumulation.
// All matrices reduce to row-major (4096 x 1024) with flat row r = s*B + b.
// f32->bf16 conversion is hoisted into one streaming pre-pass so the GEMM
// inner loops are pure global_load_b128 + v_wmma.
// ---------------------------------------------------------------------------

typedef __attribute__((ext_vector_type(16))) __bf16 v16bf;
typedef __attribute__((ext_vector_type(8)))  float  v8f;

#define Dm    1024
#define Sm    2048
#define MROWS 4096   // S*B
#define Hn    16
#define HDm   64

__device__ __forceinline__ unsigned fbits(float f) {
    union { float f; unsigned u; } x; x.f = f; return x.u;
}
// one v_perm_b32: result = { hi16(b), hi16(a) } = packed bf16 pair (truncate)
__device__ __forceinline__ unsigned pktrunc(float a, float b) {
    return __builtin_amdgcn_perm(fbits(b), fbits(a), 0x07060302u);
}
// RNE pack for epilogue stores (outside hot loops)
__device__ __forceinline__ unsigned pkbf(float a, float b) {
    const unsigned ua = fbits(a), ub = fbits(b);
    const unsigned ra = ua + 0x7FFFu + ((ua >> 16) & 1u);
    const unsigned rb = ub + 0x7FFFu + ((ub >> 16) & 1u);
    return (ra >> 16) | (rb & 0xFFFF0000u);
}
__device__ __forceinline__ unsigned short f2bf(float f) {
    return (unsigned short)pkbf(f, 0.0f);
}
__device__ __forceinline__ float bf2f(unsigned short u) {
    union { unsigned u; float f; } x; x.u = ((unsigned)u) << 16;
    return x.f;
}

union BFragBits { v16bf v; uint4 q[2]; };

// A fragment (16x32 bf16, MxK): lane L holds row M=L%16; halves 0..7 = k-run
// [kb, kb+8), halves 8..15 = k-run [kb+16, kb+24), kb includes (L>=16 ? 8 : 0).
__device__ __forceinline__ v16bf a_frag_bf(const unsigned short* p, int kb) {
    BFragBits r;
    r.q[0] = *(const uint4*)(p + kb);
    r.q[1] = *(const uint4*)(p + kb + 16);
    return r.v;
}
// B fragment (32x16 bf16, KxN): lane L holds col N=L%16; halves 0..7 = k-run
// [kb, kb+8), halves 8..15 = k-run [kb+16, kb+24), kb includes (L>=16 ? 8 : 0).
__device__ __forceinline__ v16bf b_frag_bf(const unsigned short* p, int kb) {
    BFragBits r;
    r.q[0] = *(const uint4*)(p + kb);
    r.q[1] = *(const uint4*)(p + kb + 16);
    return r.v;
}
__device__ __forceinline__ v8f wmma_bf16(v16bf a, v16bf b, v8f c) {
    return __builtin_amdgcn_wmma_f32_16x16x32_bf16(
        false, a, false, b, (short)0, c, false, false);
}
__device__ __forceinline__ v8f vzero() {
    v8f z;
#pragma unroll
    for (int i = 0; i < 8; ++i) z[i] = 0.0f;
    return z;
}

// ---------------------------------------------------------------------------
// Streaming f32 -> bf16 (truncate via v_perm_b32). 8 elements / thread.
// ---------------------------------------------------------------------------
__global__ void __launch_bounds__(256)
cvt_bf16_kernel(const float* __restrict__ in, unsigned short* __restrict__ out, int n8)
{
    const int i = blockIdx.x * 256 + threadIdx.x;
    if (i >= n8) return;
    const float4 f0 = ((const float4*)in)[2 * i];
    const float4 f1 = ((const float4*)in)[2 * i + 1];
    uint4 o;
    o.x = pktrunc(f0.x, f0.y);
    o.y = pktrunc(f0.z, f0.w);
    o.z = pktrunc(f1.x, f1.y);
    o.w = pktrunc(f1.z, f1.w);
    ((uint4*)out)[i] = o;
}

// ---------------------------------------------------------------------------
// Y(4096x1024 bf16) = Xbf @ Wbf^T + bias.  transpose_out=1 stores Y^T
// (1024x4096, packed dword stores) for the V matrix.
// Block: 128 threads = 4 waves, 128x128 tile, 64x64 per wave.
// Inner loop: all 16 global_load_b128 issued up front, then 16 v_wmma drain
// them with staggered partial s_wait_loadcnt.
// ---------------------------------------------------------------------------
__global__ void __launch_bounds__(128)
proj_bf16_kernel(const unsigned short* __restrict__ Xbf,
                 const unsigned short* __restrict__ Wbf,
                 const float* __restrict__ bias, unsigned short* __restrict__ Y,
                 int transpose_out)
{
    const int lane = threadIdx.x & 31;
    const int wave = threadIdx.x >> 5;
    const int wm = wave & 1, wn = wave >> 1;
    const int m_base = blockIdx.x * 128 + wm * 64;
    const int n_base = blockIdx.y * 128 + wn * 64;
    const int l16 = lane & 15, lhi = lane >> 4;

    v8f acc[4][4];
#pragma unroll
    for (int i = 0; i < 4; ++i)
#pragma unroll
        for (int j = 0; j < 4; ++j) acc[i][j] = vzero();

    const unsigned short* arow[4];
    const unsigned short* brow[4];
#pragma unroll
    for (int t = 0; t < 4; ++t) {
        arow[t] = Xbf + (size_t)(m_base + t * 16 + l16) * Dm;
        brow[t] = Wbf + (size_t)(n_base + t * 16 + l16) * Dm;
    }

    for (int ks = 0; ks < Dm / 32; ++ks) {
        const int akb = ks * 32 + lhi * 8;
        const int bkb = ks * 32 + lhi * 8;
        if ((ks & 1) == 0) {            // bf16: 64B/iter stride, +1KB distance
#pragma unroll
            for (int t = 0; t < 4; ++t) {
                __builtin_prefetch(arow[t] + akb + 512, 0, 1);
                __builtin_prefetch(brow[t] + bkb + 512, 0, 1);
            }
        }
        v16bf a[4], b[4];
#pragma unroll
        for (int mt = 0; mt < 4; ++mt) a[mt] = a_frag_bf(arow[mt], akb);
#pragma unroll
        for (int nt = 0; nt < 4; ++nt) b[nt] = b_frag_bf(brow[nt], bkb);
#pragma unroll
        for (int nt = 0; nt < 4; ++nt)
#pragma unroll
            for (int mt = 0; mt < 4; ++mt)
                acc[mt][nt] = wmma_bf16(a[mt], b[nt], acc[mt][nt]);
    }
#pragma unroll
    for (int nt = 0; nt < 4; ++nt) {
        const int n = n_base + nt * 16 + l16;
        const float bv = bias[n];
#pragma unroll
        for (int mt = 0; mt < 4; ++mt) {
            if (transpose_out) {
                // m = m_base + mt*16 + lhi*8 + r : contiguous over r -> pack dwords
                const int m0 = m_base + mt * 16 + lhi * 8;
                unsigned* dst = (unsigned*)(Y + (size_t)n * MROWS + m0);
#pragma unroll
                for (int r = 0; r < 4; ++r)
                    dst[r] = pkbf(acc[mt][nt][2 * r] + bv, acc[mt][nt][2 * r + 1] + bv);
            } else {
#pragma unroll
                for (int r = 0; r < 8; ++r) {
                    const int m = m_base + mt * 16 + r + lhi * 8;
                    Y[(size_t)m * Dm + n] = f2bf(acc[mt][nt][r] + bv);
                }
            }
        }
    }
}

// ---------------------------------------------------------------------------
// out(4096x1024 f32) = ctx_bf16 @ Wo_bf16^T + bo
// ---------------------------------------------------------------------------
__global__ void __launch_bounds__(128)
out_proj_kernel(const unsigned short* __restrict__ Abf,
                const unsigned short* __restrict__ Wbf,
                const float* __restrict__ bias, float* __restrict__ Y)
{
    const int lane = threadIdx.x & 31;
    const int wave = threadIdx.x >> 5;
    const int wm = wave & 1, wn = wave >> 1;
    const int m_base = blockIdx.x * 128 + wm * 64;
    const int n_base = blockIdx.y * 128 + wn * 64;
    const int l16 = lane & 15, lhi = lane >> 4;

    v8f acc[4][4];
#pragma unroll
    for (int i = 0; i < 4; ++i)
#pragma unroll
        for (int j = 0; j < 4; ++j) acc[i][j] = vzero();

    const unsigned short* arow[4];
    const unsigned short* brow[4];
#pragma unroll
    for (int t = 0; t < 4; ++t) {
        arow[t] = Abf + (size_t)(m_base + t * 16 + l16) * Dm;
        brow[t] = Wbf + (size_t)(n_base + t * 16 + l16) * Dm;
    }

    for (int ks = 0; ks < Dm / 32; ++ks) {
        const int akb = ks * 32 + lhi * 8;
        const int bkb = ks * 32 + lhi * 8;
        if ((ks & 1) == 0) {
#pragma unroll
            for (int t = 0; t < 4; ++t) {
                __builtin_prefetch(arow[t] + akb + 512, 0, 1);
                __builtin_prefetch(brow[t] + bkb + 512, 0, 1);
            }
        }
        v16bf a[4], b[4];
#pragma unroll
        for (int mt = 0; mt < 4; ++mt) a[mt] = a_frag_bf(arow[mt], akb);
#pragma unroll
        for (int nt = 0; nt < 4; ++nt) b[nt] = b_frag_bf(brow[nt], bkb);
#pragma unroll
        for (int nt = 0; nt < 4; ++nt)
#pragma unroll
            for (int mt = 0; mt < 4; ++mt)
                acc[mt][nt] = wmma_bf16(a[mt], b[nt], acc[mt][nt]);
    }
#pragma unroll
    for (int nt = 0; nt < 4; ++nt) {
        const int n = n_base + nt * 16 + l16;
        const float bv = bias[n];
#pragma unroll
        for (int mt = 0; mt < 4; ++mt)
#pragma unroll
            for (int r = 0; r < 8; ++r) {
                const int m = m_base + mt * 16 + r + lhi * 8;
                Y[(size_t)m * Dm + n] = acc[mt][nt][r] + bv;
            }
    }
}

// ---------------------------------------------------------------------------
// Fused attention: one block = 32 query rows of one (batch b', head h) group.
// Phase 1: scores = Q@K^T into LDS (bf16, 32x2048 = 128 KB).
// Phase 2: softmax (scale 1/8 folded into exp), attn -> global (non-temporal),
//          normalized P -> LDS in place (bf16).
// Phase 3: ctx = P @ V  (V pre-transposed bf16) -> ctx_bf16.
// ---------------------------------------------------------------------------
__global__ void __launch_bounds__(128)
attention_kernel(const unsigned short* __restrict__ Qbf,
                 const unsigned short* __restrict__ Kbf,
                 const unsigned short* __restrict__ Vtbf,
                 float* __restrict__ attn,
                 unsigned short* __restrict__ ctxbf)
{
    extern __shared__ unsigned short sc[];          // [32][Sm] bf16 scores / P
    const int lane = threadIdx.x & 31;
    const int wave = threadIdx.x >> 5;
    const int l16 = lane & 15, lhi = lane >> 4;
    const int g  = blockIdx.y;                      // 0..31
    const int bb = g >> 4, h = g & 15;
    const int row0 = blockIdx.x * 32;               // query row within S
    const int qbase = bb * Sm + row0;               // flat row in 4096
    const int hc = h * HDm;

    // ---- Phase 1: scores. Wave w owns key columns [w*512, w*512+512). ----
    v16bf qa[2][2];
#pragma unroll
    for (int mt = 0; mt < 2; ++mt) {
        const unsigned short* qrow = Qbf + (size_t)(qbase + mt * 16 + l16) * Dm + hc;
#pragma unroll
        for (int kk = 0; kk < 2; ++kk)
            qa[mt][kk] = a_frag_bf(qrow, kk * 32 + lhi * 8);
    }
    for (int ct = 0; ct < 32; ++ct) {
        const int n0 = wave * 512 + ct * 16;
        const unsigned short* krow = Kbf + (size_t)(bb * Sm + n0 + l16) * Dm + hc;
        v8f a0 = vzero(), a1 = vzero();
        v16bf b0 = b_frag_bf(krow, lhi * 8);
        v16bf b1 = b_frag_bf(krow, 32 + lhi * 8);
        a0 = wmma_bf16(qa[0][0], b0, a0);
        a1 = wmma_bf16(qa[1][0], b0, a1);
        a0 = wmma_bf16(qa[0][1], b1, a0);
        a1 = wmma_bf16(qa[1][1], b1, a1);
#pragma unroll
        for (int r = 0; r < 8; ++r) {
            sc[(size_t)(r + lhi * 8) * Sm      + n0 + l16] = f2bf(a0[r]);
            sc[(size_t)(16 + r + lhi * 8) * Sm + n0 + l16] = f2bf(a1[r]);
        }
    }
    __syncthreads();

    // ---- Phase 2: softmax. Wave w owns rows [w*8, w*8+8); full-wave sweeps. ----
    for (int rr = 0; rr < 8; ++rr) {
        const int row = wave * 8 + rr;
        unsigned short* rp = sc + (size_t)row * Sm;
        float mx = -3.0e38f;
        for (int i = 0; i < 32; ++i) {
            const unsigned u = *(const unsigned*)(rp + i * 64 + lane * 2);
            mx = fmaxf(mx, bf2f((unsigned short)(u & 0xffffu)));
            mx = fmaxf(mx, bf2f((unsigned short)(u >> 16)));
        }
#pragma unroll
        for (int off = 1; off < 32; off <<= 1) mx = fmaxf(mx, __shfl_xor(mx, off));
        float sum = 0.0f;
        for (int i = 0; i < 32; ++i) {
            unsigned* up = (unsigned*)(rp + i * 64 + lane * 2);
            const unsigned u = *up;
            const float e0 = __expf((bf2f((unsigned short)(u & 0xffffu)) - mx) * 0.125f);
            const float e1 = __expf((bf2f((unsigned short)(u >> 16))    - mx) * 0.125f);
            sum += e0 + e1;
            *up = pktrunc(e0, e1);                  // one v_perm_b32
        }
#pragma unroll
        for (int off = 1; off < 32; off <<= 1) sum += __shfl_xor(sum, off);
        const float inv = 1.0f / sum;
        float* arow = attn + (size_t)g * Sm * Sm + (size_t)(row0 + row) * Sm;
        for (int i = 0; i < 32; ++i) {
            unsigned* up = (unsigned*)(rp + i * 64 + lane * 2);
            const unsigned u = *up;
            const float p0 = bf2f((unsigned short)(u & 0xffffu)) * inv;
            const float p1 = bf2f((unsigned short)(u >> 16)) * inv;
            __builtin_nontemporal_store(p0, arow + i * 64 + lane * 2);      // attn never re-read
            __builtin_nontemporal_store(p1, arow + i * 64 + lane * 2 + 1);
            *up = pktrunc(p0, p1);
        }
    }
    __syncthreads();

    // ---- Phase 3: ctx = P @ V. Wave w: m-tile (w&1), n-tiles {2*(w>>1), +1}. ----
    const int mt = wave & 1;
    const int np = wave >> 1;
    v8f c0 = vzero(), c1 = vzero();
    const unsigned short* prow  = sc + (size_t)(mt * 16 + l16) * Sm;
    const unsigned short* v0row = Vtbf + (size_t)(hc + np * 32 + l16) * MROWS + bb * Sm;
    const unsigned short* v1row = v0row + (size_t)16 * MROWS;
    for (int ks = 0; ks < Sm / 32; ++ks) {
        const int kb = ks * 32;
        if ((ks & 1) == 0) {                           // bf16: +1KB ahead
            __builtin_prefetch(v0row + kb + 512, 0, 1);
            __builtin_prefetch(v1row + kb + 512, 0, 1);
        }
        v16bf a  = a_frag_bf(prow,  kb + lhi * 8);
        v16bf b0 = b_frag_bf(v0row, kb + lhi * 8);
        v16bf b1 = b_frag_bf(v1row, kb + lhi * 8);
        c0 = wmma_bf16(a, b0, c0);
        c1 = wmma_bf16(a, b1, c1);
    }
#pragma unroll
    for (int r = 0; r < 8; ++r) {
        const size_t m = (size_t)(qbase + mt * 16 + r + lhi * 8);
        ctxbf[m * Dm + hc + np * 32      + l16] = f2bf(c0[r]);
        ctxbf[m * Dm + hc + np * 32 + 16 + l16] = f2bf(c1[r]);
    }
}

// ---------------------------------------------------------------------------
extern "C" void kernel_launch(void* const* d_in, const int* in_sizes, int n_in,
                              void* d_out, int out_size, void* d_ws, size_t ws_size,
                              hipStream_t stream)
{
    const float* query = (const float*)d_in[0];
    const float* key   = (const float*)d_in[1];
    const float* value = (const float*)d_in[2];
    const float* Wq = (const float*)d_in[3];
    const float* bq = (const float*)d_in[4];
    const float* Wk = (const float*)d_in[5];
    const float* bk = (const float*)d_in[6];
    const float* Wv = (const float*)d_in[7];
    const float* bv = (const float*)d_in[8];
    const float* Wo = (const float*)d_in[9];
    const float* bo = (const float*)d_in[10];

    float* out  = (float*)d_out;
    float* attn = out + (size_t)MROWS * Dm;   // outputs: out (S,B,D) then attn (B,H,S,S)

    const size_t NX = (size_t)MROWS * Dm;     // 4,194,304 elements
    const size_t NW = (size_t)Dm * Dm;        // 1,048,576 elements

    // Workspace (bf16, stream-ordered reuse): total 34 MB
    unsigned short* Qbf  = (unsigned short*)d_ws;  // 8 MB
    unsigned short* Kbf  = Qbf  + NX;              // 8 MB
    unsigned short* Vtbf = Kbf  + NX;              // 8 MB (transposed 1024x4096)
    unsigned short* Xbf  = Vtbf + NX;              // 8 MB staging; later reused as ctx
    unsigned short* Wbf  = Xbf  + NX;              // 2 MB staging (Wq/Wk/Wv/Wo in turn)
    unsigned short* ctxbf = Xbf;                   // alias: X staging dead after proj_v

    dim3 gproj(MROWS / 128, Dm / 128);
    const int gx = (int)(NX / 8 / 256);            // 2048 blocks
    const int gw = (int)(NW / 8 / 256);            // 512 blocks

    // Q projection
    cvt_bf16_kernel<<<gx, 256, 0, stream>>>(query, Xbf, (int)(NX / 8));
    cvt_bf16_kernel<<<gw, 256, 0, stream>>>(Wq, Wbf, (int)(NW / 8));
    proj_bf16_kernel<<<gproj, 128, 0, stream>>>(Xbf, Wbf, bq, Qbf, 0);
    // K projection
    cvt_bf16_kernel<<<gx, 256, 0, stream>>>(key, Xbf, (int)(NX / 8));
    cvt_bf16_kernel<<<gw, 256, 0, stream>>>(Wk, Wbf, (int)(NW / 8));
    proj_bf16_kernel<<<gproj, 128, 0, stream>>>(Xbf, Wbf, bk, Kbf, 0);
    // V projection (transposed output)
    cvt_bf16_kernel<<<gx, 256, 0, stream>>>(value, Xbf, (int)(NX / 8));
    cvt_bf16_kernel<<<gw, 256, 0, stream>>>(Wv, Wbf, (int)(NW / 8));
    proj_bf16_kernel<<<gproj, 128, 0, stream>>>(Xbf, Wbf, bv, Vtbf, 1);

    // Attention (ctxbf overwrites the dead X staging buffer)
    dim3 gattn(Sm / 32, 32);
    attention_kernel<<<gattn, 128, 32 * Sm * sizeof(unsigned short), stream>>>(
        Qbf, Kbf, Vtbf, attn, ctxbf);

    // Output projection
    cvt_bf16_kernel<<<gw, 256, 0, stream>>>(Wo, Wbf, (int)(NW / 8));
    out_proj_kernel<<<gproj, 128, 0, stream>>>(ctxbf, Wbf, bo, out);
}
